// Model_46909632807724
// MI455X (gfx1250) — compile-verified, matrix operand order checked by reading, not run on hardware
//
#include <hip/hip_runtime.h>
#include <cstdint>

// ---------------- problem constants (from reference) ----------------
#define HID   300
#define DD    433
#define AD    133
#define BD    147
#define NA    51201      // atoms (incl. pad row 0)
#define NBND  102401     // bonds (incl. pad row 0)
#define NMOL  1024
#define APM   50
#define NSEQ  51200      // NMOL*APM
#define MAXNB 6

// padded sizes for WMMA tiling (M mult of 128, K mult of 32, N mult of 128)
#define MPA     51328    // 401*128
#define MPB     102528   // 801*128
#define KP_FEAT 160      // 133/147 -> 160
#define KP_H    320      // 300 -> 320
#define KP_D    448      // 433 -> 448
#define NP_H    384      // 300 -> 3 N-tiles of 128
#define NP_D3   1408     // 1299 -> 11 N-tiles of 128
#define LDGI    1312     // row stride for 1299-wide fp32 outputs
#define ND3     1299     // 3*433

typedef __attribute__((ext_vector_type(16))) __bf16 v16bf;
typedef __attribute__((ext_vector_type(8)))  __bf16 v8bf;
typedef __attribute__((ext_vector_type(8)))  float  v8f;

// ---- async global->LDS staging (gfx1250 path), with sync fallback ----
#if defined(__AMDGCN__) && __has_builtin(__builtin_amdgcn_global_load_async_to_lds_b128)
#define HAVE_ASYNC_LDS 1
// builtin prototype is (v4i addrspace(1)*, v4i addrspace(3)*, imm, imm)
// with v4i = gcc-style int __vector_size__(16)
typedef int v4i_g __attribute__((vector_size(16)));
typedef __attribute__((address_space(1))) v4i_g* gptr_v4i;
typedef __attribute__((address_space(3))) v4i_g* lptr_v4i;
__device__ __forceinline__ void async_copy16(const void* g, void* l)
{
  __builtin_amdgcn_global_load_async_to_lds_b128(
      (gptr_v4i)(uintptr_t)g,
      (lptr_v4i)(uint32_t)(uintptr_t)l, 0, 0);
}
__device__ __forceinline__ void async_wait0()
{
#if __has_builtin(__builtin_amdgcn_s_wait_asynccnt)
  __builtin_amdgcn_s_wait_asynccnt(0);
#else
  asm volatile("s_wait_asynccnt 0x0" ::: "memory");
#endif
}
#endif

__device__ __forceinline__ v16bf ldfrag(const __bf16* p)
{
  v8bf lo = *(const v8bf*)p;
  v8bf hi = *(const v8bf*)(p + 16);
  return __builtin_shufflevector(lo, hi,
      0,1,2,3,4,5,6,7,8,9,10,11,12,13,14,15);
}

// =====================================================================
// WMMA GEMM:  out[M,Nreal] = f( X[M,Kp] * Wt[Np,Kp]^T + bias + addMat )
// X row-major bf16 (K-padded, M-padded w/ zeros), Wt = weight transposed
// to [N,K] row-major bf16 (padded).  256 threads = 8 waves arranged 4x2:
// block tile 128(M) x 128(N); each wave 32x64 -> 2 A-frags x 4 B-frags =
// 8 WMMAs per 32-wide K step.  B tile (128x32, 8KB) staged to LDS with
// gfx1250 async-to-LDS copies (ASYNCcnt) when available.
// =====================================================================
__global__ __launch_bounds__(256) void wmma_gemm_k(
    const __bf16* __restrict__ X, int ldX,
    const __bf16* __restrict__ Wt, int ldW,
    float* __restrict__ out, int ldOut,
    const float* __restrict__ addMat, int ldAdd,
    const float* __restrict__ bias,
    int M, int Nreal, int Kp, int doRelu)
{
  __shared__ __align__(16) __bf16 Bsh[128 * 32];

  const int wave = threadIdx.x >> 5;
  const int lane = threadIdx.x & 31;
  const int sub  = lane & 15;          // row within fragment
  const int hih  = lane >> 4;          // K-half select
  const int wm   = wave >> 1;          // 0..3
  const int wn   = wave & 1;           // 0..1
  const int m0   = blockIdx.x * 128 + wm * 32;
  const int n0   = blockIdx.y * 128 + wn * 64;

  const int sn = threadIdx.x >> 2;       // 0..63 (staging row)
  const int sk = (threadIdx.x & 3) << 3; // 0,8,16,24 (staging K)
  const int nb0 = blockIdx.y * 128;

  v8f acc[2][4] = {};

  for (int kc = 0; kc < Kp; kc += 32) {
    __syncthreads();                   // previous tile fully consumed
    {
      const __bf16* g0 = &Wt[(size_t)(nb0 + sn)      * ldW + kc + sk];
      const __bf16* g1 = &Wt[(size_t)(nb0 + sn + 64) * ldW + kc + sk];
      __bf16* l0 = &Bsh[sn        * 32 + sk];
      __bf16* l1 = &Bsh[(sn + 64) * 32 + sk];
#ifdef HAVE_ASYNC_LDS
      async_copy16(g0, l0);
      async_copy16(g1, l1);
      async_wait0();
#else
      *(v8bf*)l0 = *(const v8bf*)g0;
      *(v8bf*)l1 = *(const v8bf*)g1;
#endif
    }
    __syncthreads();                   // staged tile visible to all waves

    const __bf16* arow0 = X + (size_t)(m0 + sub) * ldX + kc + hih * 8;
    const __bf16* arow1 = arow0 + (size_t)16 * ldX;
    __builtin_prefetch(arow0 + 32, 0, 0);   // next K chunk -> global_prefetch
    v16bf a0 = ldfrag(arow0);
    v16bf a1 = ldfrag(arow1);

#pragma unroll
    for (int j = 0; j < 4; ++j) {
      const __bf16* brow = &Bsh[(wn * 64 + j * 16 + sub) * 32 + hih * 8];
      v16bf b = ldfrag(brow);
      acc[0][j] = __builtin_amdgcn_wmma_f32_16x16x32_bf16(
          false, a0, false, b, (short)0, acc[0][j], false, false);
      acc[1][j] = __builtin_amdgcn_wmma_f32_16x16x32_bf16(
          false, a1, false, b, (short)0, acc[1][j], false, false);
    }
  }

  // epilogue: D element (m,n): lane -> n = n0+j*16+sub, vgpr r -> m = m0+i*16+8*hih+r
#pragma unroll
  for (int i = 0; i < 2; ++i) {
#pragma unroll
    for (int j = 0; j < 4; ++j) {
      int n = n0 + j * 16 + sub;
      if (n >= Nreal) continue;
      float bv = bias ? bias[n] : 0.0f;
#pragma unroll
      for (int r = 0; r < 8; ++r) {
        int m = m0 + i * 16 + hih * 8 + r;
        if (m >= M) continue;
        float v = acc[i][j][r] + bv;
        if (addMat) v += addMat[(size_t)m * ldAdd + n];
        if (doRelu) v = fmaxf(v, 0.0f);
        out[(size_t)m * ldOut + n] = v;
      }
    }
  }
}

// ---------------- conversion / packing kernels ----------------
// weight [K,N] fp32 -> transposed padded bf16 [Np,Kp]
__global__ void cvt_wT_k(const float* __restrict__ W, __bf16* __restrict__ Wt,
                         int K, int N, int Kp, int Np)
{
  long i = (long)blockIdx.x * blockDim.x + threadIdx.x;
  long tot = (long)Np * Kp;
  if (i >= tot) return;
  int n = (int)(i / Kp), k = (int)(i % Kp);
  float v = (k < K && n < N) ? W[(size_t)k * N + n] : 0.0f;
  Wt[i] = (__bf16)v;
}

// activations [M,K] fp32 (row stride ldX) -> padded bf16 [Mp,Kp]
__global__ void cvt_rows_k(const float* __restrict__ X, __bf16* __restrict__ Y,
                           int M, int K, int Mp, int Kp, int ldX)
{
  long i = (long)blockIdx.x * blockDim.x + threadIdx.x;
  long tot = (long)Mp * Kp;
  if (i >= tot) return;
  int m = (int)(i / Kp), k = (int)(i % Kp);
  float v = (m < M && k < K) ? X[(size_t)m * ldX + k] : 0.0f;
  Y[i] = (__bf16)v;
}

// ---------------- message-passing kernels ----------------
// agg[a,h] = sum_j msg[a2b[a,j],h] * max_j msg[a2b[a,j],h]
__global__ void aggregate_k(const float* __restrict__ msg,
                            const int* __restrict__ a2b,
                            float* __restrict__ agg)
{
  long i = (long)blockIdx.x * blockDim.x + threadIdx.x;
  if (i >= (long)NA * HID) return;
  int a = (int)(i / HID), h = (int)(i % HID);
  const int* nb = a2b + (size_t)a * MAXNB;
  float s = 0.0f, mx = -3.4e38f;
#pragma unroll
  for (int j = 0; j < MAXNB; ++j) {
    float v = msg[(size_t)nb[j] * HID + h];
    s += v;
    mx = fmaxf(mx, v);
  }
  agg[i] = s * mx;
}

// Xb[b,k] = bf16( agg[b2a[b],k] - msg[b2revb[b],k] ), zero padded
__global__ void bond_x_k(const float* __restrict__ agg,
                         const float* __restrict__ msg,
                         const int* __restrict__ b2a,
                         const int* __restrict__ b2revb,
                         __bf16* __restrict__ Xb)
{
  long i = (long)blockIdx.x * blockDim.x + threadIdx.x;
  long tot = (long)MPB * KP_H;
  if (i >= tot) return;
  int b = (int)(i / KP_H), k = (int)(i % KP_H);
  float v = 0.0f;
  if (b < NBND && k < HID)
    v = agg[(size_t)b2a[b] * HID + k] - msg[(size_t)b2revb[b] * HID + k];
  Xb[i] = (__bf16)v;
}

// node = [agg*input_atom | f_atoms] ; message = relu(node + gru_bias)
__global__ void node_msg_k(const float* __restrict__ agg,
                           const float* __restrict__ input_atom,
                           const float* __restrict__ f_atoms,
                           const float* __restrict__ gru_bias,
                           float* __restrict__ node,
                           float* __restrict__ message)
{
  long i = (long)blockIdx.x * blockDim.x + threadIdx.x;
  if (i >= (long)NA * DD) return;
  int a = (int)(i / DD), c = (int)(i % DD);
  float v = (c < HID) ? agg[(size_t)a * HID + c] * input_atom[(size_t)a * HID + c]
                      : f_atoms[(size_t)a * AD + (c - HID)];
  node[i] = v;
  message[i] = fmaxf(v + gru_bias[c], 0.0f);
}

// h0[mol,c] = max over atoms of node
__global__ void h0_k(const float* __restrict__ node, float* __restrict__ h)
{
  long i = (long)blockIdx.x * blockDim.x + threadIdx.x;
  if (i >= (long)NMOL * DD) return;
  int mol = (int)(i / DD), c = (int)(i % DD);
  float mx = -3.4e38f;
  for (int a = 0; a < APM; ++a)
    mx = fmaxf(mx, node[(size_t)(1 + mol * APM + a) * DD + c]);
  h[i] = mx;
}

__device__ __forceinline__ float sigmoidf(float x) {
  return 1.0f / (1.0f + __expf(-x));
}

// fused GRU gates; gi precomputed for all steps in GI; writes h (fp32),
// hBF (next-step GEMM input, bf16 padded) and the output row into message.
__global__ void gru_gate_k(const float* __restrict__ GI,
                           const float* __restrict__ gh,
                           float* __restrict__ h,
                           __bf16* __restrict__ hBF,
                           float* __restrict__ message, int t)
{
  long i = (long)blockIdx.x * blockDim.x + threadIdx.x;
  if (i >= (long)NMOL * DD) return;
  int mol = (int)(i / DD), c = (int)(i % DD);
  size_t gir = (size_t)(mol * APM + t) * LDGI;
  size_t ghr = (size_t)mol * LDGI;
  float r = sigmoidf(GI[gir + c]          + gh[ghr + c]);
  float z = sigmoidf(GI[gir + DD + c]     + gh[ghr + DD + c]);
  float n = tanhf(GI[gir + 2 * DD + c] + r * gh[ghr + 2 * DD + c]);
  float hold = h[i];
  float hn = (1.0f - z) * n + z * hold;
  h[i] = hn;
  hBF[(size_t)mol * KP_D + c] = (__bf16)hn;           // pads stay zero
  message[(size_t)(1 + mol * APM + t) * DD + c] = hn;
}

// mol_vec[mol,h] = max over atoms of atom_hiddens
__global__ void molmax_k(const float* __restrict__ ah, float* __restrict__ mv)
{
  long i = (long)blockIdx.x * blockDim.x + threadIdx.x;
  if (i >= (long)NMOL * HID) return;
  int mol = (int)(i / HID), hc = (int)(i % HID);
  float mx = -3.4e38f;
  for (int a = 0; a < APM; ++a)
    mx = fmaxf(mx, ah[(size_t)(1 + mol * APM + a) * HID + hc]);
  mv[i] = mx;
}

// regression head: out[mol] = ffnh[mol,:] . w2 + b2
__global__ void head_k(const float* __restrict__ ffnh,
                       const float* __restrict__ w2,
                       const float* __restrict__ b2,
                       float* __restrict__ out)
{
  int mol = blockIdx.x * blockDim.x + threadIdx.x;
  if (mol >= NMOL) return;
  float s = b2[0];
  for (int k = 0; k < HID; ++k)
    s += ffnh[(size_t)mol * HID + k] * w2[k];
  out[mol] = s;
}

// =====================================================================
// host side
// =====================================================================
static inline int nblk(long n) { return (int)((n + 255) / 256); }

static void gemm(hipStream_t s, const __bf16* X, int ldX,
                 const __bf16* Wt, int ldW, float* out, int ldOut,
                 const float* add, int ldAdd, const float* bias,
                 int Mp, int M, int Nreal, int Kp, int relu)
{
  dim3 g(Mp / 128, (Nreal + 127) / 128), b(256);
  wmma_gemm_k<<<g, b, 0, s>>>(X, ldX, Wt, ldW, out, ldOut,
                              add, ldAdd, bias, M, Nreal, Kp, relu);
}

extern "C" void kernel_launch(void* const* d_in, const int* in_sizes, int n_in,
                              void* d_out, int out_size, void* d_ws, size_t ws_size,
                              hipStream_t stream)
{
  (void)in_sizes; (void)n_in; (void)out_size; (void)ws_size;
  const float* f_atoms  = (const float*)d_in[0];
  const float* f_bonds  = (const float*)d_in[1];
  const float* Wi_atom  = (const float*)d_in[2];
  const float* Wi_bond  = (const float*)d_in[3];
  const float* Wh_bond  = (const float*)d_in[4];
  const float* Wo_w     = (const float*)d_in[5];
  const float* Wo_b     = (const float*)d_in[6];
  const float* gru_bias = (const float*)d_in[7];
  const float* gru_Wih  = (const float*)d_in[8];
  const float* gru_Whh  = (const float*)d_in[9];
  const float* gru_bih  = (const float*)d_in[10];
  const float* gru_bhh  = (const float*)d_in[11];
  const float* ffn_w1   = (const float*)d_in[12];
  const float* ffn_b1   = (const float*)d_in[13];
  const float* ffn_w2   = (const float*)d_in[14];
  const float* ffn_b2   = (const float*)d_in[15];
  const int*   a2b      = (const int*)d_in[16];
  const int*   b2a      = (const int*)d_in[17];
  const int*   b2revb   = (const int*)d_in[18];
  float* out = (float*)d_out;

  // --- workspace carve-up ---
  char* ws = (char*)d_ws;
  size_t off = 0;
  auto take = [&](size_t bytes) -> void* {
    void* p = ws + off;
    off = (off + bytes + 255) & ~(size_t)255;
    return p;
  };
  __bf16* WiAT  = (__bf16*)take((size_t)NP_H  * KP_FEAT * 2);
  __bf16* WiBT  = (__bf16*)take((size_t)NP_H  * KP_FEAT * 2);
  __bf16* WhT   = (__bf16*)take((size_t)NP_H  * KP_H   * 2);
  __bf16* WoT   = (__bf16*)take((size_t)NP_H  * KP_D   * 2);
  __bf16* WihT  = (__bf16*)take((size_t)NP_D3 * KP_D   * 2);
  __bf16* WhhT  = (__bf16*)take((size_t)NP_D3 * KP_D   * 2);
  __bf16* W1T   = (__bf16*)take((size_t)NP_H  * KP_H   * 2);
  __bf16* fA16  = (__bf16*)take((size_t)MPA * KP_FEAT * 2);
  __bf16* fB16  = (__bf16*)take((size_t)MPB * KP_FEAT * 2);
  float*  inAtom  = (float*)take((size_t)NA   * HID * 4);
  float*  inBond  = (float*)take((size_t)NBND * HID * 4);
  float*  msgA    = (float*)take((size_t)NBND * HID * 4);
  float*  msgB    = (float*)take((size_t)NBND * HID * 4);
  float*  agg     = (float*)take((size_t)NA   * HID * 4);
  __bf16* Xb16    = (__bf16*)take((size_t)MPB * KP_H * 2);
  float*  node    = (float*)take((size_t)NA * DD * 4);
  float*  message = (float*)take((size_t)NA * DD * 4);
  __bf16* seq16   = (__bf16*)take((size_t)NSEQ * KP_D * 2);
  float*  GI      = (float*)take((size_t)NSEQ * LDGI * 4);
  float*  gh      = (float*)take((size_t)NMOL * LDGI * 4);
  float*  h       = (float*)take((size_t)NMOL * DD * 4);
  __bf16* h16     = (__bf16*)take((size_t)NMOL * KP_D * 2);
  __bf16* msg16   = (__bf16*)take((size_t)MPA * KP_D * 2);
  float*  ah      = (float*)take((size_t)NA * HID * 4);
  float*  molvec  = (float*)take((size_t)NMOL * HID * 4);
  __bf16* mv16    = (__bf16*)take((size_t)NMOL * KP_H * 2);
  float*  ffnh    = (float*)take((size_t)NMOL * HID * 4);

  // --- weight packing (transpose + bf16 + pad) ---
  cvt_wT_k<<<nblk((long)NP_H  * KP_FEAT), 256, 0, stream>>>(Wi_atom, WiAT, AD,  HID, KP_FEAT, NP_H);
  cvt_wT_k<<<nblk((long)NP_H  * KP_FEAT), 256, 0, stream>>>(Wi_bond, WiBT, BD,  HID, KP_FEAT, NP_H);
  cvt_wT_k<<<nblk((long)NP_H  * KP_H  ), 256, 0, stream>>>(Wh_bond, WhT,  HID, HID, KP_H,    NP_H);
  cvt_wT_k<<<nblk((long)NP_H  * KP_D  ), 256, 0, stream>>>(Wo_w,    WoT,  DD,  HID, KP_D,    NP_H);
  cvt_wT_k<<<nblk((long)NP_D3 * KP_D  ), 256, 0, stream>>>(gru_Wih, WihT, DD,  ND3, KP_D,    NP_D3);
  cvt_wT_k<<<nblk((long)NP_D3 * KP_D  ), 256, 0, stream>>>(gru_Whh, WhhT, DD,  ND3, KP_D,    NP_D3);
  cvt_wT_k<<<nblk((long)NP_H  * KP_H  ), 256, 0, stream>>>(ffn_w1,  W1T,  HID, HID, KP_H,    NP_H);

  // --- feature conversion ---
  cvt_rows_k<<<nblk((long)MPA * KP_FEAT), 256, 0, stream>>>(f_atoms, fA16, NA,   AD, MPA, KP_FEAT, AD);
  cvt_rows_k<<<nblk((long)MPB * KP_FEAT), 256, 0, stream>>>(f_bonds, fB16, NBND, BD, MPB, KP_FEAT, BD);

  // --- input projections: relu(X @ Wi) ---
  gemm(stream, fA16, KP_FEAT, WiAT, KP_FEAT, inAtom, HID, nullptr, 0, nullptr,
       MPA, NA, HID, KP_FEAT, 1);
  gemm(stream, fB16, KP_FEAT, WiBT, KP_FEAT, inBond, HID, nullptr, 0, nullptr,
       MPB, NBND, HID, KP_FEAT, 1);

  // --- message passing: msg starts at input_bond ---
  const float* msgCur = inBond;
  float* dstA = msgA; float* dstB = msgB;
  for (int it = 0; it < 5; ++it) {          // DEPTH-1
    aggregate_k<<<nblk((long)NA * HID), 256, 0, stream>>>(msgCur, a2b, agg);
    bond_x_k<<<nblk((long)MPB * KP_H), 256, 0, stream>>>(agg, msgCur, b2a, b2revb, Xb16);
    gemm(stream, Xb16, KP_H, WhT, KP_H, dstA, HID, inBond, HID, nullptr,
         MPB, NBND, HID, KP_H, 1);          // relu(input_bond + X@Wh)
    msgCur = dstA;
    float* t = dstA; dstA = dstB; dstB = t;
  }
  aggregate_k<<<nblk((long)NA * HID), 256, 0, stream>>>(msgCur, a2b, agg);

  // --- node / message / h0 ---
  node_msg_k<<<nblk((long)NA * DD), 256, 0, stream>>>(agg, inAtom, f_atoms, gru_bias,
                                                      node, message);
  h0_k<<<nblk((long)NMOL * DD), 256, 0, stream>>>(node, h);

  // --- batch all GRU input-side GEMMs: GI = seq @ Wih + bih ---
  cvt_rows_k<<<nblk((long)NSEQ * KP_D), 256, 0, stream>>>(message + DD, seq16,
                                                          NSEQ, DD, NSEQ, KP_D, DD);
  gemm(stream, seq16, KP_D, WihT, KP_D, GI, LDGI, nullptr, 0, gru_bih,
       NSEQ, NSEQ, ND3, KP_D, 0);
  cvt_rows_k<<<nblk((long)NMOL * KP_D), 256, 0, stream>>>(h, h16, NMOL, DD, NMOL, KP_D, DD);

  // --- sequential GRU scan (only gh depends on h) ---
  for (int t = 0; t < APM; ++t) {
    gemm(stream, h16, KP_D, WhhT, KP_D, gh, LDGI, nullptr, 0, gru_bhh,
         NMOL, NMOL, ND3, KP_D, 0);
    gru_gate_k<<<nblk((long)NMOL * DD), 256, 0, stream>>>(GI, gh, h, h16, message, t);
  }

  // --- output projection: relu(message @ Wo + b) ---
  cvt_rows_k<<<nblk((long)MPA * KP_D), 256, 0, stream>>>(message, msg16, NA, DD, MPA, KP_D, DD);
  gemm(stream, msg16, KP_D, WoT, KP_D, ah, HID, nullptr, 0, Wo_b,
       MPA, NA, HID, KP_D, 1);

  // --- per-mol max pool + FFN head ---
  molmax_k<<<nblk((long)NMOL * HID), 256, 0, stream>>>(ah, molvec);
  cvt_rows_k<<<nblk((long)NMOL * KP_H), 256, 0, stream>>>(molvec, mv16, NMOL, HID, NMOL, KP_H, HID);
  gemm(stream, mv16, KP_H, W1T, KP_H, ffnh, HID, nullptr, 0, ffn_b1,
       NMOL, NMOL, HID, KP_H, 1);
  head_k<<<nblk(NMOL), 256, 0, stream>>>(ffnh, ffn_w2, ffn_b2, out);
}
